// Pyramid_11123965297249
// MI455X (gfx1250) — matrix operand order repaired and checked by reading, not compiled, not run on hardware
//
#include <hip/hip_runtime.h>

// ---------------------------------------------------------------------------
// MI455X (gfx1250): bf16 WMMA GEMMs, async LDS gather pipeline, L2-resident x.
// ---------------------------------------------------------------------------

typedef __attribute__((ext_vector_type(16))) __bf16       v16bf;
typedef __attribute__((ext_vector_type(8)))  float        v8f;
typedef __attribute__((ext_vector_type(4)))  unsigned int v4u;

union Frag { v16bf bf; v4u u[2]; };

#define NPTS   200000
#define CIN    256
#define COUT4  64
#define KOFF   27
#define LDSROW 264          // 256 + 8 pad (halves) -> bank-conflict-free b128
#define LDSBUF (16 * LDSROW)
#define EPSBN  1e-5f

// float -> bf16, round-to-nearest-even
__device__ __forceinline__ unsigned short f2bf(float f) {
  unsigned u = __builtin_bit_cast(unsigned, f);
  unsigned r = 0x7FFFu + ((u >> 16) & 1u);
  return (unsigned short)((u + r) >> 16);
}

// ---------------------------------------------------------------------------
// x [N,256] fp32 -> xb [(N+1),256] bf16 (row N is the zero "missing" row)
// ---------------------------------------------------------------------------
__global__ void k_pack_x(const float* __restrict__ x, unsigned short* __restrict__ xb,
                         long nxc, long total) {
  long i = (long)blockIdx.x * blockDim.x + threadIdx.x;
  if (i >= total) return;
  xb[i] = (i < nxc) ? f2bf(x[i]) : (unsigned short)0;
}

// ---------------------------------------------------------------------------
// Pack W [nSlices, Kdim, Nc] fp32 (row-major) into the WMMA B-fragment layout:
//   per (kb, nbi): 32 lanes x 8 dwords; lane = {h=lane>>4, n=(lane&15)+nbi*16}
//   dword d = { W[kb*32 + h*8 + 2*(d&3) + 16*(d>>2), n] , W[..+1, n] } as packed bf16.
// ---------------------------------------------------------------------------
__global__ void k_pack_w(const float* __restrict__ W, unsigned int* __restrict__ Wp,
                         int Kdim, int Nc, int nSlices) {
  int per = (Kdim >> 5) * (Nc >> 4) * 256;   // dwords per slice
  long i = (long)blockIdx.x * blockDim.x + threadIdx.x;
  if (i >= (long)per * nSlices) return;
  int s = (int)(i / per), j = (int)(i % per);
  int d = j & 7, lane = (j >> 3) & 31, t = j >> 8;
  int nb  = Nc >> 4;
  int nbi = t % nb, kb = t / nb;
  int h = lane >> 4, n = (lane & 15) + (nbi << 4);
  int k = (kb << 5) + (h << 3) + ((d & 3) << 1) + ((d >> 2) << 4);
  const float* Ws = W + (long)s * Kdim * Nc;
  unsigned lo = f2bf(Ws[(long)k * Nc + n]);
  unsigned hi = f2bf(Ws[(long)(k + 1) * Nc + n]);
  Wp[i] = lo | (hi << 16);
}

// ---------------------------------------------------------------------------
// Gather-GEMM: z[16rows, 64cols] = sum_k  xb[nbr[:,k]] @ W[k]  (+bias)
// 4 waves/block, one 16x16 output tile per wave, bf16 WMMA 16x16x32.
// Double-buffered async gather: GLOBAL_LOAD_ASYNC_TO_LDS_B128 for offset k+1
// is in flight while the 8 WMMAs of offset k execute (ASYNCcnt pipeline).
// nbr == nullptr -> identity gather (plain 1x1 conv).
// ---------------------------------------------------------------------------
__global__ __launch_bounds__(128)
void k_gemm_gather(const unsigned short* __restrict__ xb,
                   const int* __restrict__ nbr,
                   const unsigned int* __restrict__ Wp,
                   const float* __restrict__ bias,
                   float* __restrict__ ztmp, float* __restrict__ stats,
                   int N, int Koff) {
  __shared__ __align__(16) unsigned short As[2 * LDSBUF];

  const int row0 = blockIdx.x << 4;
  const int t    = threadIdx.x;
  const int wv   = t >> 5;        // wave = output 16-col tile (0..3)
  const int lane = t & 31;
  const int m    = lane & 15;
  const int h    = lane >> 4;
  const int r    = t >> 3;        // gather: 8 threads per row
  const int seg  = t & 7;         // 64B segment within the 512B row

  // issue async gather of 16 rows (64B/thread) for offset k into buffer buf
  auto issue_gather = [&](int k, int buf) {
    int idx = nbr ? nbr[(long)(row0 + r) * Koff + k] : (row0 + r);
    const unsigned short* g = xb + (long)idx * CIN + seg * 32;
    unsigned ldsa = (unsigned)(size_t)(As + buf * LDSBUF + r * LDSROW + seg * 32);
    // INST_OFFSET is added to both the LDS and the global address (ISA 10.7)
    asm volatile(
        "global_load_async_to_lds_b128 %0, %1, off\n\t"
        "global_load_async_to_lds_b128 %0, %1, off offset:16\n\t"
        "global_load_async_to_lds_b128 %0, %1, off offset:32\n\t"
        "global_load_async_to_lds_b128 %0, %1, off offset:48"
        :: "v"(ldsa), "v"(g) : "memory");
  };

  issue_gather(0, 0);
  asm volatile("s_wait_asynccnt 0" ::: "memory");
  __syncthreads();

  v8f acc = {};
  for (int k = 0; k < Koff; ++k) {
    const int buf = k & 1;
    if (k + 1 < Koff) issue_gather(k + 1, buf ^ 1);   // overlap with WMMA below

    const unsigned int* wk = Wp + (long)k * 8192;     // 8 kb * 4 nb * 256 dwords
    if (k + 1 < Koff) __builtin_prefetch(wk + 8192, 0, 1);
    const unsigned short* Ab = As + buf * LDSBUF;
#pragma unroll
    for (int kb = 0; kb < 8; ++kb) {
      Frag a, b;
      // A frag per ISA layout: lane<16 K{0..7,16..23}, lane>=16 K{8..15,24..31}
      a.u[0] = *(const v4u*)(Ab + m * LDSROW + kb * 32 + h * 8);
      a.u[1] = *(const v4u*)(Ab + m * LDSROW + kb * 32 + 16 + h * 8);
      const v4u* bp = (const v4u*)(wk + (((kb << 2) + wv) * 32 + lane) * 8);
      b.u[0] = bp[0]; b.u[1] = bp[1];
      acc = __builtin_amdgcn_wmma_f32_16x16x32_bf16(
          false, a.bf, false, b.bf, (short)0, acc, false, false);
    }
    if (k + 1 < Koff)
      asm volatile("s_wait_asynccnt 0" ::: "memory");  // next buffer landed
    __syncthreads();
  }

  // Epilogue: bias, store pre-BN fp32, reduce BN stats.
  const int col = (wv << 4) + m;
  const float bv = bias[col];
  float ps = 0.f, pq = 0.f;
#pragma unroll
  for (int i = 0; i < 8; ++i) {
    float z = acc[i] + bv;
    int row = row0 + h * 8 + i;
    ztmp[(long)row * COUT4 + col] = z;
    ps += z; pq += z * z;
  }
  ps += __shfl_xor(ps, 16);
  pq += __shfl_xor(pq, 16);
  if (h == 0) {
    atomicAdd(&stats[col], ps);
    atomicAdd(&stats[COUT4 + col], pq);
  }
}

// ---------------------------------------------------------------------------
// BN(train) + ReLU + bf16 into concat buffer column slice.
// ---------------------------------------------------------------------------
__global__ void k_bn_store(const float* __restrict__ z, const float* __restrict__ stats,
                           const float* __restrict__ g, const float* __restrict__ be,
                           unsigned short* __restrict__ xcat, int N, int colOff) {
  long i = (long)blockIdx.x * blockDim.x + threadIdx.x;
  if (i >= (long)N * COUT4) return;
  int c = (int)(i & 63); long row = i >> 6;
  float mean = stats[c] / (float)N;
  float var  = stats[COUT4 + c] / (float)N - mean * mean;
  float s  = g[c] * rsqrtf(var + EPSBN);
  float sh = be[c] - mean * s;
  float v = fmaxf(z[i] * s + sh, 0.f);
  xcat[row * 320 + colOff + c] = f2bf(v);
}

// ---------------------------------------------------------------------------
// Segment sums per batch (batch_idx sorted -> run-length accumulate, rare atomics)
// ---------------------------------------------------------------------------
__global__ void k_pool_sum(const float* __restrict__ x, const int* __restrict__ bidx,
                           float* __restrict__ psum, float* __restrict__ pcnt,
                           int N, int rpb) {
  int c = threadIdx.x;                         // 256 threads = channels
  int r0 = blockIdx.x * rpb;
  if (r0 >= N) return;
  int r1 = r0 + rpb; if (r1 > N) r1 = N;
  float acc = 0.f; int cnt = 0; int cb = bidx[r0];
  for (int r = r0; r < r1; ++r) {
    int b = bidx[r];
    if (b != cb) {
      atomicAdd(&psum[cb * CIN + c], acc);
      if (c == 0) atomicAdd(&pcnt[cb], (float)cnt);
      acc = 0.f; cnt = 0; cb = b;
    }
    acc += x[(long)r * CIN + c]; cnt++;
  }
  atomicAdd(&psum[cb * CIN + c], acc);
  if (c == 0) atomicAdd(&pcnt[cb], (float)cnt);
}

// p5[8,64] = (psum/cnt) @ W5 + b5 ; count-weighted BN stats over broadcast rows
__global__ void k_pool_fc(const float* __restrict__ psum, const float* __restrict__ pcnt,
                          const float* __restrict__ W5, const float* __restrict__ b5,
                          float* __restrict__ p5, float* __restrict__ stats5, int N) {
  int c = threadIdx.x;                         // 0..63
  float ssum = 0.f, ssq = 0.f;
  for (int b = 0; b < 8; ++b) {
    float cnt = fmaxf(pcnt[b], 1.f);
    float inv = 1.f / cnt;
    float z = b5[c];
    for (int k = 0; k < CIN; ++k)
      z += (psum[b * CIN + k] * inv) * W5[k * COUT4 + c];
    p5[b * COUT4 + c] = z;
    ssum += cnt * z; ssq += cnt * z * z;
  }
  stats5[c] = ssum; stats5[COUT4 + c] = ssq;
}

__global__ void k_bn_store_pool(const float* __restrict__ p5, const float* __restrict__ stats5,
                                const int* __restrict__ bidx,
                                const float* __restrict__ g, const float* __restrict__ be,
                                unsigned short* __restrict__ xcat, int N) {
  long i = (long)blockIdx.x * blockDim.x + threadIdx.x;
  if (i >= (long)N * COUT4) return;
  int c = (int)(i & 63); long row = i >> 6;
  float mean = stats5[c] / (float)N;
  float var  = stats5[COUT4 + c] / (float)N - mean * mean;
  float s  = g[c] * rsqrtf(var + EPSBN);
  float sh = be[c] - mean * s;
  float v = fmaxf(p5[(long)bidx[row] * COUT4 + c] * s + sh, 0.f);
  xcat[row * 320 + 256 + c] = f2bf(v);
}

// ---------------------------------------------------------------------------
// Final GEMM: xcat [N,320] bf16 @ Wc [320,256].  phase 0: BN stats only.
// phase 1: recompute (33 GFLOP), normalize + ReLU + residual -> d_out fp32.
// ---------------------------------------------------------------------------
__global__ __launch_bounds__(128)
void k_gemm_cat(const unsigned short* __restrict__ xcat,
                const unsigned int* __restrict__ Wp,
                const float* __restrict__ bias,
                const float* __restrict__ g, const float* __restrict__ be,
                const float* __restrict__ xres, float* __restrict__ out,
                float* __restrict__ stats, int N, int phase) {
  const int row0 = blockIdx.x << 4;
  const int t = threadIdx.x;
  const int wv = t >> 5, lane = t & 31, m = lane & 15, h = lane >> 4;
  const int nbi = blockIdx.y * 4 + wv;         // 0..15 col tiles
  v8f acc = {};
#pragma unroll
  for (int kb = 0; kb < 10; ++kb) {
    Frag a, b;
    const unsigned short* ar = xcat + (long)(row0 + m) * 320 + kb * 32 + h * 8;
    a.u[0] = *(const v4u*)ar;
    a.u[1] = *(const v4u*)(ar + 16);
    const unsigned int* bp = Wp + (((kb << 4) + nbi) * 32 + lane) * 8;
    b.u[0] = *(const v4u*)bp; b.u[1] = *(const v4u*)(bp + 4);
    acc = __builtin_amdgcn_wmma_f32_16x16x32_bf16(
        false, a.bf, false, b.bf, (short)0, acc, false, false);
  }
  const int col = (nbi << 4) + m;
  const float bv = bias[col];
  if (phase == 0) {
    float ps = 0.f, pq = 0.f;
#pragma unroll
    for (int i = 0; i < 8; ++i) { float z = acc[i] + bv; ps += z; pq += z * z; }
    ps += __shfl_xor(ps, 16);
    pq += __shfl_xor(pq, 16);
    if (h == 0) {
      atomicAdd(&stats[col], ps);
      atomicAdd(&stats[CIN + col], pq);
    }
  } else {
    float mean = stats[col] / (float)N;
    float var  = stats[CIN + col] / (float)N - mean * mean;
    float s  = g[col] * rsqrtf(var + EPSBN);
    float sh = be[col] - mean * s;
#pragma unroll
    for (int i = 0; i < 8; ++i) {
      int row = row0 + h * 8 + i;
      float z = fmaxf((acc[i] + bv) * s + sh, 0.f);
      out[(long)row * CIN + col] = z + xres[(long)row * CIN + col];
    }
  }
}

// ---------------------------------------------------------------------------
extern "C" void kernel_launch(void* const* d_in, const int* in_sizes, int n_in,
                              void* d_out, int out_size, void* d_ws, size_t ws_size,
                              hipStream_t stream) {
  const int N = NPTS;
  const float* x     = (const float*)d_in[0];
  const int*   nbr6  = (const int*)d_in[1];
  const int*   nbr12 = (const int*)d_in[2];
  const int*   nbr18 = (const int*)d_in[3];
  const int*   bidx  = (const int*)d_in[4];
  const float* W1 = (const float*)d_in[5];  const float* b1 = (const float*)d_in[6];
  const float* W6 = (const float*)d_in[7];  const float* b6 = (const float*)d_in[8];
  const float* W12= (const float*)d_in[9];  const float* b12= (const float*)d_in[10];
  const float* W18= (const float*)d_in[11]; const float* b18= (const float*)d_in[12];
  const float* W5 = (const float*)d_in[13]; const float* b5 = (const float*)d_in[14];
  const float* Wc = (const float*)d_in[15]; const float* bc = (const float*)d_in[16];
  const float* g1 = (const float*)d_in[17]; const float* be1= (const float*)d_in[18];
  const float* g2 = (const float*)d_in[19]; const float* be2= (const float*)d_in[20];
  const float* g3 = (const float*)d_in[21]; const float* be3= (const float*)d_in[22];
  const float* g4 = (const float*)d_in[23]; const float* be4= (const float*)d_in[24];
  const float* g5 = (const float*)d_in[25]; const float* be5= (const float*)d_in[26];
  const float* gcw= (const float*)d_in[27]; const float* becw=(const float*)d_in[28];
  float* out = (float*)d_out;

  // --- workspace layout ---
  char* ws = (char*)d_ws;
  float* stats  = (float*)ws;            // 5 branches x [sum64|sumsq64] = 640 f
  float* statsC = stats + 640;           // 512 f (sum256|sumsq256)
  float* psum   = statsC + 512;          // 8*256
  float* pcnt   = psum + 2048;           // 8
  float* p5     = pcnt + 8;              // 8*64
  size_t off = 16384;
  unsigned short* xb   = (unsigned short*)(ws + off); off += (size_t)(N + 1) * CIN * 2;
  unsigned short* xcat = (unsigned short*)(ws + off); off += (size_t)N * 320 * 2;
  float* ztmp = (float*)(ws + off);                   off += (size_t)N * COUT4 * 4;
  unsigned int* W1p  = (unsigned int*)(ws + off);     off += 8192u * 4;
  unsigned int* W6p  = (unsigned int*)(ws + off);     off += (size_t)KOFF * 8192 * 4;
  unsigned int* W12p = (unsigned int*)(ws + off);     off += (size_t)KOFF * 8192 * 4;
  unsigned int* W18p = (unsigned int*)(ws + off);     off += (size_t)KOFF * 8192 * 4;
  unsigned int* Wcp  = (unsigned int*)(ws + off);     off += 40960u * 4;

  hipMemsetAsync(d_ws, 0, 16384, stream);   // zero stats / pool accumulators

  // pack activations + weights into bf16 / B-fragment layouts
  {
    long total = (long)(N + 1) * CIN;
    k_pack_x<<<(unsigned)((total + 255) / 256), 256, 0, stream>>>(x, xb, (long)N * CIN, total);
  }
  k_pack_w<<<(8192 + 255) / 256, 256, 0, stream>>>(W1, W1p, 256, 64, 1);
  k_pack_w<<<(KOFF * 8192 + 255) / 256, 256, 0, stream>>>(W6,  W6p,  256, 64, KOFF);
  k_pack_w<<<(KOFF * 8192 + 255) / 256, 256, 0, stream>>>(W12, W12p, 256, 64, KOFF);
  k_pack_w<<<(KOFF * 8192 + 255) / 256, 256, 0, stream>>>(W18, W18p, 256, 64, KOFF);
  k_pack_w<<<(40960 + 255) / 256, 256, 0, stream>>>(Wc, Wcp, 320, 256, 1);

  // pool sums (independent of GEMMs)
  k_pool_sum<<<(N + 511) / 512, 256, 0, stream>>>(x, bidx, psum, pcnt, N, 512);

  const unsigned nelems = (unsigned)(((long)N * COUT4 + 255) / 256);

  // branch 1: x @ W1
  k_gemm_gather<<<N / 16, 128, 0, stream>>>(xb, nullptr, W1p, b1, ztmp, stats + 0, N, 1);
  k_bn_store<<<nelems, 256, 0, stream>>>(ztmp, stats + 0, g1, be1, xcat, N, 0);
  // branches 2-4: sparse convs
  k_gemm_gather<<<N / 16, 128, 0, stream>>>(xb, nbr6,  W6p,  b6,  ztmp, stats + 128, N, KOFF);
  k_bn_store<<<nelems, 256, 0, stream>>>(ztmp, stats + 128, g2, be2, xcat, N, 64);
  k_gemm_gather<<<N / 16, 128, 0, stream>>>(xb, nbr12, W12p, b12, ztmp, stats + 256, N, KOFF);
  k_bn_store<<<nelems, 256, 0, stream>>>(ztmp, stats + 256, g3, be3, xcat, N, 128);
  k_gemm_gather<<<N / 16, 128, 0, stream>>>(xb, nbr18, W18p, b18, ztmp, stats + 384, N, KOFF);
  k_bn_store<<<nelems, 256, 0, stream>>>(ztmp, stats + 384, g4, be4, xcat, N, 192);
  // branch 5: pooled
  k_pool_fc<<<1, 64, 0, stream>>>(psum, pcnt, W5, b5, p5, stats + 512, N);
  k_bn_store_pool<<<nelems, 256, 0, stream>>>(p5, stats + 512, bidx, g5, be5, xcat, N);

  // final concat GEMM: stats pass, then fused BN+ReLU+residual pass
  dim3 grd(N / 16, 4);
  k_gemm_cat<<<grd, 128, 0, stream>>>(xcat, Wcp, bc, gcw, becw, x, out, statsC, N, 0);
  k_gemm_cat<<<grd, 128, 0, stream>>>(xcat, Wcp, bc, gcw, becw, x, out, statsC, N, 1);
}
